// SelfAttention_46428596470146
// MI455X (gfx1250) — matrix operand order repaired and checked by reading, not compiled, not run on hardware
//
#include <hip/hip_runtime.h>
#include <math.h>

typedef __attribute__((ext_vector_type(16))) __bf16 v16bf;
typedef __attribute__((ext_vector_type(8)))  __bf16 v8bf;
typedef __attribute__((ext_vector_type(4)))  __bf16 v4bf;
typedef __attribute__((ext_vector_type(8)))  float  v8f;

constexpr int B = 32, S = 4096, U = 512;
constexpr int MTILE   = 64;        // seq rows per workgroup (4 x 16 M-subtiles)
constexpr int NCHUNK  = 32;        // output cols per pass  (2 x 16 N-subtiles)
constexpr int NCHUNKS = U / NCHUNK;
constexpr int KBLK    = U / 32;    // 16 WMMA K-steps per chunk
constexpr int LDS_K   = U + 8;     // +8 bf16 pad -> row stride = 260 dwords = 4 mod 64 banks

// ---------------- workspace layout (bytes) ----------------
// base    [B][U]  f32      @ 0        (64 KB)
// UwT     [U][U]  bf16     @ 65536    (512 KB)  (UwT[v][k] = Uw[k][v])
// logits  [B][S]  f32      @ 589824   (512 KB)
// partial [B][16][U] f32   @ 1114112  (1 MB)

__global__ __launch_bounds__(512) void base_kernel(
    const float* __restrict__ s_prev, const float* __restrict__ Ww,
    const float* __restrict__ Wb, const float* __restrict__ Ub,
    float* __restrict__ base) {
  const int b = blockIdx.x, v = threadIdx.x;
  const float* sp = s_prev + b * U;
  float acc = Wb[v] + Ub[v];
  for (int k = 0; k < U; ++k) acc += sp[k] * Ww[k * U + v];
  base[b * U + v] = acc;
}

__global__ __launch_bounds__(256) void uwt_kernel(const float* __restrict__ Uw,
                                                  __bf16* __restrict__ UwT) {
  const int k = blockIdx.x, t = threadIdx.x;
  for (int v = t; v < U; v += 256) UwT[(size_t)v * U + k] = (__bf16)Uw[(size_t)k * U + v];
}

// Async DMA one Uw chunk (NCHUNK rows x U bf16 = 32 KB) global -> padded LDS rows.
// GLOBAL_LOAD_ASYNC_TO_LDS_B128, GVS mode; tracked on ASYNCcnt (no VGPR data path).
__device__ __forceinline__ void stage_uw_async(const __bf16* __restrict__ usrc,
                                               __bf16* dst, int t) {
  const unsigned long long sbase = (unsigned long long)usrc;
  #pragma unroll
  for (int i = 0; i < (NCHUNK * U * 2 / 16) / 256; ++i) {   // 8 x b128 per thread
    const int q  = t + 256 * i;          // b128 id; 64 per 512-bf16 row
    const int n  = q >> 6;
    const int k8 = (q & 63) << 3;        // bf16 element offset within row
    const unsigned lds_off = (unsigned)(unsigned long long)(const void*)(dst + n * LDS_K + k8);
    const unsigned goff    = (unsigned)(q * 16);
    asm volatile("global_load_async_to_lds_b128 %0, %1, %2"
                 :: "v"(lds_off), "v"(goff), "s"(sbase)
                 : "memory");
  }
}

__device__ __forceinline__ void wait_async0() {
  asm volatile("s_wait_asynccnt 0x0" ::: "memory");
}

// Fused: logits[b][s] = tanh(base[b][:] + hidden[b][s][:]@Uw) . Vw + Vb
// - bf16 WMMA, f32 accumulate; [B,S,U] intermediate never touches HBM (saves 512 MB)
// - Uw B-tiles double-buffered in LDS via async-to-LDS DMA, overlapped with WMMA
// - fragment registers double-buffered so ds_load latency hides under v_wmma
__global__ __launch_bounds__(256) void logits_kernel(
    const float* __restrict__ hidden, const __bf16* __restrict__ UwT,
    const float* __restrict__ base, const float* __restrict__ Vw,
    const float* __restrict__ Vb, float* __restrict__ logits) {
  extern __shared__ char smem[];
  __bf16* lds_h  = (__bf16*)smem;                                  // [MTILE][LDS_K]
  __bf16* lds_u0 = (__bf16*)(smem + MTILE * LDS_K * 2);            // [NCHUNK][LDS_K]
  __bf16* lds_u1 = lds_u0 + NCHUNK * LDS_K;                        // [NCHUNK][LDS_K]
  float*  lds_l  = (float*)(smem + (MTILE + 2 * NCHUNK) * LDS_K * 2); // [MTILE]

  const int t    = threadIdx.x;
  const int wave = t >> 5, lane = t & 31;
  const int half = lane >> 4, lc = lane & 15;
  const int wave_m = wave & 3;    // 4 M sub-tiles of 16 rows
  const int wave_n = wave >> 2;   // 2 N sub-tiles of 16 cols
  const int b  = blockIdx.y;
  const int s0 = blockIdx.x * MTILE;

  if (t < MTILE) lds_l[t] = 0.0f;

  // Kick off async DMA of Uw chunk 0 while we VALU-convert the hidden strip.
  stage_uw_async(UwT, lds_u0, t);

  // Stage hidden strip [MTILE x U] f32 -> bf16 LDS (conversion => must go through VGPRs).
  const float* hsrc = hidden + ((size_t)b * S + s0) * U;
  #pragma unroll
  for (int i = 0; i < (MTILE * U / 4) / 256; ++i) {
    const int q = t + 256 * i;          // float4 id; 128 per row
    const int row = q >> 7;
    const int c4  = (q & 127) << 2;
    const float4 f = *(const float4*)(hsrc + (size_t)row * U + c4);
    v4bf h; h.x = (__bf16)f.x; h.y = (__bf16)f.y; h.z = (__bf16)f.z; h.w = (__bf16)f.w;
    *(v4bf*)(lds_h + row * LDS_K + c4) = h;
  }
  wait_async0();
  __syncthreads();

  float acc[8];
  #pragma unroll
  for (int r = 0; r < 8; ++r) acc[r] = 0.0f;

  const __bf16* arow = lds_h + (wave_m * 16 + lc) * LDS_K;

  for (int nc = 0; nc < NCHUNKS; ++nc) {
    __bf16* cur = (nc & 1) ? lds_u1 : lds_u0;
    __bf16* nxt = (nc & 1) ? lds_u0 : lds_u1;
    // DMA next chunk into the other buffer; overlaps with the 16 WMMAs below.
    if (nc + 1 < NCHUNKS)
      stage_uw_async(UwT + (size_t)(nc + 1) * NCHUNK * U, nxt, t);

    const __bf16* brow = cur + (wave_n * 16 + lc) * LDS_K;

    // ISA 16-bit A layout: lane half h needs K {8h..8h+7, 16+8h..16+8h+7};
    // B layout: same K order as A. Fragments double-buffered in regs.
    v8bf alo = *(const v8bf*)(arow + 8 * half);
    v8bf ahi = *(const v8bf*)(arow + 16 + 8 * half);
    v8bf blo = *(const v8bf*)(brow + 8 * half);
    v8bf bhi = *(const v8bf*)(brow + 16 + 8 * half);

    v8f c = {0.f, 0.f, 0.f, 0.f, 0.f, 0.f, 0.f, 0.f};
    #pragma unroll
    for (int kb = 0; kb < KBLK; ++kb) {
      v16bf a  = __builtin_shufflevector(alo, ahi, 0,1,2,3,4,5,6,7,8,9,10,11,12,13,14,15);
      v16bf bm = __builtin_shufflevector(blo, bhi, 0,1,2,3,4,5,6,7,8,9,10,11,12,13,14,15);
      if (kb + 1 < KBLK) {                       // prefetch next K-step's fragments
        const int ko = (kb + 1) * 32;
        alo = *(const v8bf*)(arow + ko + 8 * half);
        ahi = *(const v8bf*)(arow + ko + 16 + 8 * half);
        blo = *(const v8bf*)(brow + ko + 8 * half);
        bhi = *(const v8bf*)(brow + ko + 16 + 8 * half);
      }
      c = __builtin_amdgcn_wmma_f32_16x16x32_bf16(false, a, false, bm, (short)0, c,
                                                  false, false);
    }

    // Epilogue: tanh(U_h + base) * Vw, butterfly-reduce over the 16 column lanes.
    const int v  = nc * NCHUNK + wave_n * 16 + lc;
    const float bc = base[b * U + v];
    const float vw = Vw[v];
    float tv[8];
    #pragma unroll
    for (int r = 0; r < 8; ++r) tv[r] = tanhf(c[r] + bc) * vw;
    #pragma unroll
    for (int m = 1; m < 16; m <<= 1) {
      #pragma unroll
      for (int r = 0; r < 8; ++r) tv[r] += __shfl_xor(tv[r], m, 32);
    }
    #pragma unroll
    for (int r = 0; r < 8; ++r) acc[r] += tv[r];

    wait_async0();        // next chunk's DMA must have landed before anyone reads it
    __syncthreads();
  }

  // C layout: VGPR r, lanes 0-15 -> row r, lanes 16-31 -> row r+8 (within 16-subtile).
  const int rbase = wave_m * 16 + half * 8;
  if (lc == 0) {
    #pragma unroll
    for (int r = 0; r < 8; ++r) atomicAdd(&lds_l[rbase + r], acc[r]);
  }
  __syncthreads();
  if (t < MTILE) logits[(size_t)b * S + s0 + t] = lds_l[t] + Vb[0];
}

__global__ __launch_bounds__(256) void softmax_kernel(const float* __restrict__ logits,
                                                      float* __restrict__ weights) {
  __shared__ float red[256];
  const int b = blockIdx.x, t = threadIdx.x;
  const float* lrow = logits + (size_t)b * S;
  float m = -INFINITY;
  for (int s = t; s < S; s += 256) m = fmaxf(m, lrow[s]);
  red[t] = m; __syncthreads();
  for (int o = 128; o > 0; o >>= 1) { if (t < o) red[t] = fmaxf(red[t], red[t + o]); __syncthreads(); }
  m = red[0]; __syncthreads();
  float sum = 0.f;
  for (int s = t; s < S; s += 256) sum += __expf(lrow[s] - m);
  red[t] = sum; __syncthreads();
  for (int o = 128; o > 0; o >>= 1) { if (t < o) red[t] += red[t + o]; __syncthreads(); }
  const float inv = 1.0f / red[0];
  float* wrow = weights + (size_t)b * S;
  for (int s = t; s < S; s += 256) wrow[s] = __expf(lrow[s] - m) * inv;
}

__global__ __launch_bounds__(256) void ctx_partial_kernel(
    const float* __restrict__ hidden, const float* __restrict__ weights,
    float* __restrict__ partial) {
  const int chunk = blockIdx.x, b = blockIdx.y, t = threadIdx.x;
  const int SC = S / 16;
  const float* w = weights + (size_t)b * S + (size_t)chunk * SC;
  const float* h = hidden + ((size_t)b * S + (size_t)chunk * SC) * U;
  float a0 = 0.f, a1 = 0.f;
  for (int s = 0; s < SC; ++s) {
    __builtin_prefetch(h + (size_t)(s + 8) * U + t, 0, 0);   // global_prefetch_b8
    const float ws = w[s];
    a0 += ws * h[(size_t)s * U + t];
    a1 += ws * h[(size_t)s * U + t + 256];
  }
  float* p = partial + ((size_t)b * 16 + chunk) * U;
  p[t] = a0; p[t + 256] = a1;
}

__global__ __launch_bounds__(512) void ctx_reduce_kernel(const float* __restrict__ partial,
                                                         float* __restrict__ ctx) {
  const int b = blockIdx.x, t = threadIdx.x;
  float a = 0.f;
  #pragma unroll
  for (int c = 0; c < 16; ++c) a += partial[((size_t)b * 16 + c) * U + t];
  ctx[b * U + t] = a;
}

extern "C" void kernel_launch(void* const* d_in, const int* in_sizes, int n_in,
                              void* d_out, int out_size, void* d_ws, size_t ws_size,
                              hipStream_t stream) {
  const float* s_prev = (const float*)d_in[0];
  const float* hidden = (const float*)d_in[1];
  const float* Ww     = (const float*)d_in[2];
  const float* Wb     = (const float*)d_in[3];
  const float* Uw     = (const float*)d_in[4];
  const float* Ub     = (const float*)d_in[5];
  const float* Vw     = (const float*)d_in[6];
  const float* Vb     = (const float*)d_in[7];

  float* out     = (float*)d_out;
  float* ctx     = out;            // [B][U]
  float* weights = out + B * U;    // [B][S] (tuple order: context, weights)

  char*   ws      = (char*)d_ws;
  float*  base    = (float*)(ws);
  __bf16* UwT     = (__bf16*)(ws + 65536);
  float*  logits  = (float*)(ws + 589824);
  float*  partial = (float*)(ws + 1114112);

  base_kernel<<<B, U, 0, stream>>>(s_prev, Ww, Wb, Ub, base);
  uwt_kernel<<<U, 256, 0, stream>>>(Uw, UwT);

  // LDS: hidden 66560 + 2x Uw 66560 + 256 = 133376 B  (2 workgroups / 320 KB WGP)
  const size_t smem = (size_t)(MTILE + 2 * NCHUNK) * LDS_K * 2 + MTILE * 4;
  logits_kernel<<<dim3(S / MTILE, B), 256, smem, stream>>>(hidden, UwT, base, Vw, Vb, logits);

  softmax_kernel<<<B, 256, 0, stream>>>(logits, weights);
  ctx_partial_kernel<<<dim3(16, B), 256, 0, stream>>>(hidden, weights, partial);
  ctx_reduce_kernel<<<B, 512, 0, stream>>>(partial, ctx);
}
